// Encoders_13451837571792
// MI455X (gfx1250) — matrix operand order repaired and checked by reading, compiled, not run on hardware
//
#include <hip/hip_runtime.h>

// ---------------- types / helpers ----------------
typedef __bf16 bf16t;
typedef __attribute__((ext_vector_type(16))) __bf16 v16bf;
typedef __attribute__((ext_vector_type(8)))  __bf16 v8bf;
typedef __attribute__((ext_vector_type(8)))  float  v8f;

__device__ __forceinline__ float bf2f(bf16t x) {
    unsigned int u = ((unsigned int)__builtin_bit_cast(unsigned short, x)) << 16;
    return __builtin_bit_cast(float, u);
}
__device__ __forceinline__ bf16t f2bf(float f) {
    unsigned int u = __builtin_bit_cast(unsigned int, f);
    unsigned short r = (unsigned short)((u + 0x7FFFu + ((u >> 16) & 1u)) >> 16);
    return __builtin_bit_cast(bf16t, r);
}

// A-fragment (16x32 bf16): lane l -> row m=l%16; 16 contiguous K values
// starting at (l/16)*16.  p must point at lds_row_base + (lane>>4)*8.
__device__ __forceinline__ v16bf ldfragA(const bf16t* p) {
    const int hs = ((threadIdx.x & 31) >> 4) * 8;
    v8bf lo = *(const v8bf*)(p + hs);
    v8bf hi = *(const v8bf*)(p + hs + 8);
    v16bf r;
#pragma unroll
    for (int i = 0; i < 8; ++i) { r[i] = lo[i]; r[8 + i] = hi[i]; }
    return r;
}
// B-fragment (32x16 bf16): lane l -> col n=l%16, 16 contiguous K values
// starting at (l<16?0:16).  p points at K-major row for this n, + kb.
__device__ __forceinline__ v16bf ldfragB(const bf16t* p) {
    v8bf lo = *(const v8bf*)(p);
    v8bf hi = *(const v8bf*)(p + 8);
    v16bf r;
#pragma unroll
    for (int i = 0; i < 8; ++i) { r[i] = lo[i]; r[8 + i] = hi[i]; }
    return r;
}
__device__ __forceinline__ v8f wmma_bf16(v16bf a, v16bf b, v8f c) {
    return __builtin_amdgcn_wmma_f32_16x16x32_bf16(false, a, false, b, (short)0, c,
                                                   false, false);
}

// ---------------- constants ----------------
#define CB   4
#define CS   1024
#define CDM  512
#define CH   8
#define CD   64
#define CDFF 2048
#define MROWS (CB * CS)   // 4096

// ---------------- small kernels ----------------
__global__ void k_f2b(const float* __restrict__ s, bf16t* __restrict__ d, int n) {
    int i = blockIdx.x * 256 + threadIdx.x;
    if (i < n) d[i] = f2bf(s[i]);
}

__global__ void k_init_nz(const int* __restrict__ protok, float* __restrict__ nzf) {
    __shared__ int red[256];
    int t = threadIdx.x, c = 0;
    for (int i = t; i < CS; i += 256) c += (protok[i] != 0) ? 1 : 0;
    red[t] = c; __syncthreads();
    for (int s = 128; s > 0; s >>= 1) { if (t < s) red[t] += red[t + s]; __syncthreads(); }
    if (t == 0) nzf[0] = (float)red[0];
}

__global__ void k_zero_sums(float* __restrict__ sums) {
    if (threadIdx.x < CB * CH) sums[threadIdx.x] = 0.0f;
}

// y = LN(a + t0) * g + be ; also bf16 copy
__global__ __launch_bounds__(256) void k_add_ln(const float* __restrict__ a,
                                                const float* __restrict__ t0,
                                                const float* __restrict__ g,
                                                const float* __restrict__ be,
                                                float* __restrict__ y,
                                                bf16t* __restrict__ yb) {
    __shared__ float red[256];
    const int row = blockIdx.x, t = threadIdx.x;
    const size_t base = (size_t)row * CDM;
    float v0 = a[base + t] + t0[base + t];
    float v1 = a[base + t + 256] + t0[base + t + 256];
    red[t] = v0 + v1; __syncthreads();
    for (int s = 128; s > 0; s >>= 1) { if (t < s) red[t] += red[t + s]; __syncthreads(); }
    const float mean = red[0] * (1.0f / CDM);
    __syncthreads();
    const float d0 = v0 - mean, d1 = v1 - mean;
    red[t] = d0 * d0 + d1 * d1; __syncthreads();
    for (int s = 128; s > 0; s >>= 1) { if (t < s) red[t] += red[t + s]; __syncthreads(); }
    const float r = rsqrtf(red[0] * (1.0f / CDM) + 1e-9f);
    float o0 = d0 * r * g[t] + be[t];
    float o1 = d1 * r * g[t + 256] + be[t + 256];
    y[base + t] = o0;       y[base + t + 256] = o1;
    yb[base + t] = f2bf(o0); yb[base + t + 256] = f2bf(o1);
}

// ---------------- generic bf16 GEMM: C = A[MxK] * Bw[KxN] + bias ----------------
// block tile 128x128, K step 32, 8 waves (2x4), each wave 64x32 (8 WMMA tiles).
// Compile-time epilogue config -> straight-line stores, no per-element branches.
template <int RELU, int HEADS, int HASF, int HASB>
__global__ __launch_bounds__(256) void k_gemm(const bf16t* __restrict__ A,
                                              const bf16t* __restrict__ Bw,
                                              const float* __restrict__ bias,
                                              float* __restrict__ outF,
                                              bf16t* __restrict__ outB,
                                              int M, int N, int K) {
    __shared__ __align__(16) bf16t As[128 * 40];
    __shared__ __align__(16) bf16t Bt[128 * 40];   // transposed: [n][k]
    const int tid = threadIdx.x, lane = tid & 31, wid = tid >> 5;
    const int wm = wid & 1, wn = wid >> 1;
    const int m0 = blockIdx.y * 128, n0 = blockIdx.x * 128;
    const int lrow = tid >> 1, lhalf = tid & 1;    // A loader: 128 rows x 32
    const int brow = tid >> 3, bseg = tid & 7;     // B loader: 32 k-rows x 128

    v8f acc[4][2];
#pragma unroll
    for (int mi = 0; mi < 4; ++mi)
#pragma unroll
        for (int ni = 0; ni < 2; ++ni)
#pragma unroll
            for (int r = 0; r < 8; ++r) acc[mi][ni][r] = 0.0f;

    const bf16t* gpa = A + (size_t)(m0 + lrow) * K + lhalf * 16;
    const bf16t* gpb = Bw + (size_t)brow * N + n0 + bseg * 16;

    for (int k0 = 0; k0 < K; k0 += 32) {
        const v8bf* ga = (const v8bf*)(gpa + k0);
        v8bf a0 = ga[0], a1 = ga[1];
        *(v8bf*)(&As[lrow * 40 + lhalf * 16])     = a0;
        *(v8bf*)(&As[lrow * 40 + lhalf * 16 + 8]) = a1;
        const v8bf* gb = (const v8bf*)(gpb + (size_t)k0 * N);
        v8bf b0 = gb[0], b1 = gb[1];
#pragma unroll
        for (int e = 0; e < 8; ++e) {
            Bt[(bseg * 16 + e) * 40 + brow]     = b0[e];
            Bt[(bseg * 16 + 8 + e) * 40 + brow] = b1[e];
        }
        // hint next tiles into cache while WMMAs run (global_prefetch_b8)
        if (k0 + 32 < K) {
            __builtin_prefetch((const void*)(gpa + k0 + 32), 0, 1);
            __builtin_prefetch((const void*)(gpb + (size_t)(k0 + 32) * N), 0, 1);
        }
        __syncthreads();
        v16bf af[4], bfr[2];
        const int aoff = (lane >> 4) * 8;
#pragma unroll
        for (int mi = 0; mi < 4; ++mi)
            af[mi] = ldfragA(&As[(wm * 64 + mi * 16 + (lane & 15)) * 40 + aoff]);
        const int kb = (lane < 16) ? 0 : 16;
#pragma unroll
        for (int ni = 0; ni < 2; ++ni)
            bfr[ni] = ldfragB(&Bt[(wn * 32 + ni * 16 + (lane & 15)) * 40 + kb]);
#pragma unroll
        for (int mi = 0; mi < 4; ++mi)
#pragma unroll
            for (int ni = 0; ni < 2; ++ni)
                acc[mi][ni] = wmma_bf16(af[mi], bfr[ni], acc[mi][ni]);
        __syncthreads();
    }

#pragma unroll
    for (int mi = 0; mi < 4; ++mi) {
#pragma unroll
        for (int ni = 0; ni < 2; ++ni) {
            const int col = n0 + wn * 32 + ni * 16 + (lane & 15);
            const float bv = bias[col];
#pragma unroll
            for (int r = 0; r < 8; ++r) {
                const int row = m0 + wm * 64 + mi * 16 + r + 8 * (lane >> 4);
                float v = acc[mi][ni][r] + bv;
                if (RELU) v = fmaxf(v, 0.0f);
                size_t idx;
                if (HEADS) {
                    const int bb = row >> 10, ss = row & 1023;
                    const int hh = col >> 6,  dd = col & 63;
                    idx = ((size_t)(bb * CH + hh) * CS + ss) * CD + dd;
                } else {
                    idx = (size_t)row * N + col;
                }
                if (HASB) outB[idx] = f2bf(v);
                if (HASF) outF[idx] = v;
            }
        }
    }
}

// ---------------- attention pass A: sum of exp(QK^T/8 + mask) per (b,h) -------
__global__ __launch_bounds__(256) void k_attn_sum(const bf16t* __restrict__ q,
                                                  const bf16t* __restrict__ k,
                                                  const float* __restrict__ mask,
                                                  float* __restrict__ sums) {
    __shared__ __align__(16) bf16t Qs[128 * 72];
    __shared__ __align__(16) bf16t Ks[128 * 72];
    __shared__ float red[256];
    const int tid = threadIdx.x, lane = tid & 31, wid = tid >> 5;
    const int wm = wid & 1, wn = wid >> 1;
    const int j0 = blockIdx.x * 128, i0 = blockIdx.y * 128, bh = blockIdx.z;
    const int b = bh >> 3;
    {
        const int row = tid >> 1, off = (tid & 1) * 32;
        const v8bf* gq = (const v8bf*)(q + ((size_t)bh * CS + i0 + row) * CD + off);
        const v8bf* gk = (const v8bf*)(k + ((size_t)bh * CS + j0 + row) * CD + off);
#pragma unroll
        for (int u = 0; u < 4; ++u) {
            *(v8bf*)(&Qs[row * 72 + off + u * 8]) = gq[u];
            *(v8bf*)(&Ks[row * 72 + off + u * 8]) = gk[u];
        }
    }
    __syncthreads();
    v8f acc[4][2];
#pragma unroll
    for (int mi = 0; mi < 4; ++mi)
#pragma unroll
        for (int ni = 0; ni < 2; ++ni)
#pragma unroll
            for (int r = 0; r < 8; ++r) acc[mi][ni][r] = 0.0f;
#pragma unroll
    for (int kt = 0; kt < 2; ++kt) {
        const int kbase = kt * 32;
        v16bf af[4], bfr[2];
#pragma unroll
        for (int mi = 0; mi < 4; ++mi)
            af[mi] = ldfragA(&Qs[(wm * 64 + mi * 16 + (lane & 15)) * 72 + kbase + (lane >> 4) * 8]);
        const int kb = kbase + ((lane < 16) ? 0 : 16);
#pragma unroll
        for (int ni = 0; ni < 2; ++ni)
            bfr[ni] = ldfragB(&Ks[(wn * 32 + ni * 16 + (lane & 15)) * 72 + kb]);
#pragma unroll
        for (int mi = 0; mi < 4; ++mi)
#pragma unroll
            for (int ni = 0; ni < 2; ++ni)
                acc[mi][ni] = wmma_bf16(af[mi], bfr[ni], acc[mi][ni]);
    }
    float lsum = 0.0f;
#pragma unroll
    for (int mi = 0; mi < 4; ++mi)
#pragma unroll
        for (int ni = 0; ni < 2; ++ni) {
            const int jj = j0 + wn * 32 + ni * 16 + (lane & 15);
#pragma unroll
            for (int r = 0; r < 8; ++r) {
                const int ii = i0 + wm * 64 + mi * 16 + r + 8 * (lane >> 4);
                const float mv = mask[((size_t)b * CS + ii) * CS + jj];
                lsum += __expf(fmaf(mv, -1e9f, acc[mi][ni][r] * 0.125f));
            }
        }
    red[tid] = lsum; __syncthreads();
    for (int s = 128; s > 0; s >>= 1) { if (tid < s) red[tid] += red[tid + s]; __syncthreads(); }
    if (tid == 0) atomicAdd(&sums[bh], red[0]);
}

// ---------------- attention pass B: O = P^T V * (nz/sum), quirk-faithful -------
// block: i-strip of 128 output rows, loops j in steps of 64 (recompute P tile)
__global__ __launch_bounds__(256) void k_attn_out(const bf16t* __restrict__ q,
                                                  const bf16t* __restrict__ k,
                                                  const bf16t* __restrict__ v,
                                                  const float* __restrict__ mask,
                                                  const float* __restrict__ sums,
                                                  const float* __restrict__ nzf,
                                                  bf16t* __restrict__ o) {
    __shared__ __align__(16) bf16t Ki[128 * 72];   // K rows i0..i0+127
    __shared__ __align__(16) bf16t QS[128 * 72];   // phase1: Q[64][72]; phase2: St[ii][jj]
    __shared__ __align__(16) bf16t Vt[64 * 72];    // V transposed: [d][jj]
    const int tid = threadIdx.x, lane = tid & 31, wid = tid >> 5;
    const int i0 = blockIdx.x * 128, bh = blockIdx.y;
    const int b = bh >> 3, h = bh & 7;
    const int swm = wid & 1, swn = wid >> 1;       // S-phase wave grid 2x4
    const int owm = wid & 3, own = wid >> 2;       // O-phase wave grid 4x2
    {
        const int row = tid >> 1, off = (tid & 1) * 32;
        const v8bf* gk = (const v8bf*)(k + ((size_t)bh * CS + i0 + row) * CD + off);
#pragma unroll
        for (int u = 0; u < 4; ++u) *(v8bf*)(&Ki[row * 72 + off + u * 8]) = gk[u];
    }
    v8f oacc[2][2];
#pragma unroll
    for (int mi = 0; mi < 2; ++mi)
#pragma unroll
        for (int ni = 0; ni < 2; ++ni)
#pragma unroll
            for (int r = 0; r < 8; ++r) oacc[mi][ni][r] = 0.0f;

    for (int j0 = 0; j0 < CS; j0 += 64) {
        __syncthreads();
        {   // load Q rows j0..j0+63 and V (transposed) rows j0..j0+63
            const int row = tid >> 2, off = (tid & 3) * 16;
            const v8bf* gq = (const v8bf*)(q + ((size_t)bh * CS + j0 + row) * CD + off);
            *(v8bf*)(&QS[row * 72 + off])     = gq[0];
            *(v8bf*)(&QS[row * 72 + off + 8]) = gq[1];
            const v8bf* gv = (const v8bf*)(v + ((size_t)bh * CS + j0 + row) * CD + off);
            v8bf v0 = gv[0], v1 = gv[1];
#pragma unroll
            for (int e = 0; e < 8; ++e) {
                Vt[(off + e) * 72 + row]     = v0[e];
                Vt[(off + 8 + e) * 72 + row] = v1[e];
            }
            if (j0 + 64 < CS) {
                __builtin_prefetch((const void*)(q + ((size_t)bh * CS + j0 + 64 + row) * CD + off), 0, 1);
                __builtin_prefetch((const void*)(v + ((size_t)bh * CS + j0 + 64 + row) * CD + off), 0, 1);
            }
        }
        __syncthreads();
        // S phase: S[64(j)][128(i)] = Q_j * K_i^T
        v8f sac[2][2];
#pragma unroll
        for (int mi = 0; mi < 2; ++mi)
#pragma unroll
            for (int ni = 0; ni < 2; ++ni)
#pragma unroll
                for (int r = 0; r < 8; ++r) sac[mi][ni][r] = 0.0f;
#pragma unroll
        for (int kt = 0; kt < 2; ++kt) {
            const int kbase = kt * 32;
            v16bf af[2], bfr[2];
#pragma unroll
            for (int mi = 0; mi < 2; ++mi)
                af[mi] = ldfragA(&QS[(swm * 32 + mi * 16 + (lane & 15)) * 72 + kbase + (lane >> 4) * 8]);
            const int kb = kbase + ((lane < 16) ? 0 : 16);
#pragma unroll
            for (int ni = 0; ni < 2; ++ni)
                bfr[ni] = ldfragB(&Ki[(swn * 32 + ni * 16 + (lane & 15)) * 72 + kb]);
#pragma unroll
            for (int mi = 0; mi < 2; ++mi)
#pragma unroll
                for (int ni = 0; ni < 2; ++ni)
                    sac[mi][ni] = wmma_bf16(af[mi], bfr[ni], sac[mi][ni]);
        }
        __syncthreads();   // everyone done reading QS (Q tile)
        // write St[ii][jj] = exp(S/8 + mask*-1e9)  (transposed store into QS)
#pragma unroll
        for (int mi = 0; mi < 2; ++mi)
#pragma unroll
            for (int ni = 0; ni < 2; ++ni) {
                const int il = swn * 32 + ni * 16 + (lane & 15);
#pragma unroll
                for (int r = 0; r < 8; ++r) {
                    const int jl = swm * 32 + mi * 16 + r + 8 * (lane >> 4);
                    const float mv = mask[((size_t)b * CS + j0 + jl) * CS + i0 + il];
                    const float e = __expf(fmaf(mv, -1e9f, sac[mi][ni][r] * 0.125f));
                    QS[il * 72 + jl] = f2bf(e);
                }
            }
        __syncthreads();
        // O phase: O[128(ii)][64(d)] += St * V   (K dim = jj, 2 steps of 32)
#pragma unroll
        for (int kt = 0; kt < 2; ++kt) {
            const int kbase = kt * 32;
            v16bf af[2], bfr[2];
#pragma unroll
            for (int mi = 0; mi < 2; ++mi)
                af[mi] = ldfragA(&QS[(owm * 32 + mi * 16 + (lane & 15)) * 72 + kbase + (lane >> 4) * 8]);
            const int kb = kbase + ((lane < 16) ? 0 : 16);
#pragma unroll
            for (int ni = 0; ni < 2; ++ni)
                bfr[ni] = ldfragB(&Vt[(own * 32 + ni * 16 + (lane & 15)) * 72 + kb]);
#pragma unroll
            for (int mi = 0; mi < 2; ++mi)
#pragma unroll
                for (int ni = 0; ni < 2; ++ni)
                    oacc[mi][ni] = wmma_bf16(af[mi], bfr[ni], oacc[mi][ni]);
        }
    }
    const float scale = nzf[0] / sums[bh];
#pragma unroll
    for (int mi = 0; mi < 2; ++mi)
#pragma unroll
        for (int ni = 0; ni < 2; ++ni) {
            const int dcol = own * 32 + ni * 16 + (lane & 15);
#pragma unroll
            for (int r = 0; r < 8; ++r) {
                const int il = owm * 32 + mi * 16 + r + 8 * (lane >> 4);
                o[((size_t)b * CS + i0 + il) * CDM + h * CD + dcol] =
                    f2bf(oacc[mi][ni][r] * scale);
            }
        }
}

// ---------------- host launcher ----------------
extern "C" void kernel_launch(void* const* d_in, const int* in_sizes, int n_in,
                              void* d_out, int out_size, void* d_ws, size_t ws_size,
                              hipStream_t stream) {
    (void)in_sizes; (void)n_in; (void)out_size; (void)ws_size;
    const float* x     = (const float*)d_in[0];
    const float* mask  = (const float*)d_in[1];
    const int*   protok= (const int*)d_in[2];
    const float* wq = (const float*)d_in[3];  const float* bq = (const float*)d_in[4];
    const float* wk = (const float*)d_in[5];  const float* bk = (const float*)d_in[6];
    const float* wv = (const float*)d_in[7];  const float* bv = (const float*)d_in[8];
    const float* wo = (const float*)d_in[9];  const float* bo = (const float*)d_in[10];
    const float* w1 = (const float*)d_in[11]; const float* b1 = (const float*)d_in[12];
    const float* w2 = (const float*)d_in[13]; const float* b2 = (const float*)d_in[14];
    const float* ln1g = (const float*)d_in[15]; const float* ln1b = (const float*)d_in[16];
    const float* ln2g = (const float*)d_in[17]; const float* ln2b = (const float*)d_in[18];
    float* out = (float*)d_out;

    char* cur = (char*)d_ws;
    auto alloc = [&](size_t bytes) { char* p = cur; cur += (bytes + 255) & ~(size_t)255; return p; };
    bf16t* hb    = (bf16t*)alloc((size_t)MROWS * CDM * 2);
    bf16t* qb    = (bf16t*)alloc((size_t)MROWS * CDM * 2);
    bf16t* kb2   = (bf16t*)alloc((size_t)MROWS * CDM * 2);
    bf16t* vb    = (bf16t*)alloc((size_t)MROWS * CDM * 2);
    bf16t* ob    = (bf16t*)alloc((size_t)MROWS * CDM * 2);
    bf16t* f1    = (bf16t*)alloc((size_t)MROWS * CDFF * 2);
    bf16t* out1b = (bf16t*)alloc((size_t)MROWS * CDM * 2);
    float* out1f = (float*)alloc((size_t)MROWS * CDM * 4);
    float* t0    = (float*)alloc((size_t)MROWS * CDM * 4);
    float* hbuf  = (float*)alloc((size_t)MROWS * CDM * 4);
    bf16t* wqb = (bf16t*)alloc((size_t)CDM * CDM * 2);
    bf16t* wkb = (bf16t*)alloc((size_t)CDM * CDM * 2);
    bf16t* wvb = (bf16t*)alloc((size_t)CDM * CDM * 2);
    bf16t* wob = (bf16t*)alloc((size_t)CDM * CDM * 2);
    bf16t* w1b = (bf16t*)alloc((size_t)CDM * CDFF * 2);
    bf16t* w2b = (bf16t*)alloc((size_t)CDFF * CDM * 2);
    float* sums = (float*)alloc(CB * CH * 4);
    float* nzf  = (float*)alloc(4);

    const int nW = CDM * CDM, nF = CDM * CDFF, nX = MROWS * CDM;
    k_f2b<<<(nW + 255) / 256, 256, 0, stream>>>(wq, wqb, nW);
    k_f2b<<<(nW + 255) / 256, 256, 0, stream>>>(wk, wkb, nW);
    k_f2b<<<(nW + 255) / 256, 256, 0, stream>>>(wv, wvb, nW);
    k_f2b<<<(nW + 255) / 256, 256, 0, stream>>>(wo, wob, nW);
    k_f2b<<<(nF + 255) / 256, 256, 0, stream>>>(w1, w1b, nF);
    k_f2b<<<(nF + 255) / 256, 256, 0, stream>>>(w2, w2b, nF);
    k_f2b<<<(nX + 255) / 256, 256, 0, stream>>>(x, hb, nX);
    k_init_nz<<<1, 256, 0, stream>>>(protok, nzf);

    const dim3 gProj(CDM / 128, MROWS / 128);
    const dim3 gFF1(CDFF / 128, MROWS / 128);

    for (int layer = 0; layer < 2; ++layer) {
        const float* hf = (layer == 0) ? x : hbuf;
        // QKV projections (head-scattered bf16 outputs)
        k_gemm<0, 1, 0, 1><<<gProj, 256, 0, stream>>>(hb, wqb, bq, nullptr, qb,
                                                      MROWS, CDM, CDM);
        k_gemm<0, 1, 0, 1><<<gProj, 256, 0, stream>>>(hb, wkb, bk, nullptr, kb2,
                                                      MROWS, CDM, CDM);
        k_gemm<0, 1, 0, 1><<<gProj, 256, 0, stream>>>(hb, wvb, bv, nullptr, vb,
                                                      MROWS, CDM, CDM);
        // attention: global exp-sum then O = P^T V * nz/sum
        k_zero_sums<<<1, 32, 0, stream>>>(sums);
        k_attn_sum<<<dim3(CS / 128, CS / 128, CB * CH), 256, 0, stream>>>(qb, kb2, mask, sums);
        k_attn_out<<<dim3(CS / 128, CB * CH), 256, 0, stream>>>(qb, kb2, vb, mask, sums, nzf, ob);
        // output projection -> t0 (f32), then LN1
        k_gemm<0, 0, 1, 0><<<gProj, 256, 0, stream>>>(ob, wob, bo, t0, nullptr,
                                                      MROWS, CDM, CDM);
        k_add_ln<<<MROWS, 256, 0, stream>>>(hf, t0, ln1g, ln1b, out1f, out1b);
        // FFN
        k_gemm<1, 0, 0, 1><<<gFF1, 256, 0, stream>>>(out1b, w1b, b1, nullptr, f1,
                                                     MROWS, CDFF, CDM);
        k_gemm<0, 0, 1, 0><<<gProj, 256, 0, stream>>>(f1, w2b, b2, t0, nullptr,
                                                      MROWS, CDM, CDFF);
        float* yout = (layer == 1) ? out : hbuf;
        k_add_ln<<<MROWS, 256, 0, stream>>>(out1f, t0, ln2g, ln2b, yout, hb);
    }
}
